// EquivariantBlock_67851893342733
// MI455X (gfx1250) — matrix operand order repaired and checked by reading, not compiled, not run on hardware
//
#include <hip/hip_runtime.h>
#include <hip/hip_bf16.h>
#include <stdint.h>

// EGNN EquivariantBlock for MI455X (gfx1250, wave32).
// All big GEMMs run on v_wmma_f32_16x16x32_bf16 with f32 accumulation.
// Tiling: 128 rows per 128-thread block; each of 4 waves owns a 32x128 tile
// (two A fragments share every B fragment -> half the LDS B traffic/FLOP).
// ws layout (~480 MB): [E,128] f32 edge buffer (reused in-place), ea/cdiff,
// agg/aggx, h_cur, BN stats, and bf16-transposed weights.

#define H       128
#define NNODES  50000
#define NEDGES  800000

typedef __attribute__((ext_vector_type(16))) __bf16 v16bf;
typedef __attribute__((ext_vector_type(8)))  float  v8f;

union FragU { v16bf v; unsigned short u[16]; };

__device__ __forceinline__ unsigned short f2bf(float f) {
  unsigned int x = __float_as_uint(f);
  x += 0x7FFFu + ((x >> 16) & 1u);          // round-to-nearest-even
  return (unsigned short)(x >> 16);
}

__device__ __forceinline__ float sigmoidf_(float x) {
  return 1.0f / (1.0f + __expf(-x));
}

// ---- WMMA fragment loads (ISA 7.12.2 wave32 layouts) ----------------------
// A (16x32 bf16): lane l -> M=l&15, koff=8*(l>>4); half j -> K=koff+(j&7)+16*(j>>3)
__device__ __forceinline__ v16bf load_frag_A(const unsigned short* lds, int rowStride,
                                             int m0, int kc) {
  const int lane = threadIdx.x & 31;
  const int m    = m0 + (lane & 15);
  const int koff = (lane >> 4) * 8;
  const unsigned short* p = lds + (size_t)m * rowStride + kc * 32;
  FragU r;
#pragma unroll
  for (int j = 0; j < 8; ++j) r.u[j]     = p[koff + j];
#pragma unroll
  for (int j = 0; j < 8; ++j) r.u[8 + j] = p[16 + koff + j];
  return r.v;
}

// B (32x16 bf16): lane l -> N=l&15, K=8*(l>>4)+(j&7)+16*(j>>3). Weights stored transposed [N][Kpad].
__device__ __forceinline__ v16bf load_frag_B(const unsigned short* ldsWT, int Kpad,
                                             int nb, int kc) {
  const int lane = threadIdx.x & 31;
  const int n  = nb * 16 + (lane & 15);
  const int k0 = kc * 32 + (lane >> 4) * 8;
  const unsigned short* p = ldsWT + (size_t)n * Kpad + k0;
  FragU r;
#pragma unroll
  for (int j = 0; j < 16; ++j) r.u[j] = p[j + (j >> 3) * 8];
  return r.v;
}

__device__ __forceinline__ void zero_acc(v8f acc[8]) {
#pragma unroll
  for (int i = 0; i < 8; ++i)
#pragma unroll
    for (int j = 0; j < 8; ++j) acc[i][j] = 0.0f;
}

// Dual-M tile: 32 rows x 128 cols per wave; both A frags share each B frag.
template <int KCHUNKS>
__device__ __forceinline__ void wmma_rowtile2(const unsigned short* ldsA, int Kpad,
                                              const unsigned short* ldsWT,
                                              int m0, v8f acc0[8], v8f acc1[8]) {
#pragma unroll
  for (int kc = 0; kc < KCHUNKS; ++kc) {
    v16bf a0 = load_frag_A(ldsA, Kpad, m0, kc);
    v16bf a1 = load_frag_A(ldsA, Kpad, m0 + 16, kc);
#pragma unroll
    for (int nb = 0; nb < 8; ++nb) {
      v16bf b = load_frag_B(ldsWT, Kpad, nb, kc);
      acc0[nb] = __builtin_amdgcn_wmma_f32_16x16x32_bf16(
          false, a0, false, b, (short)0, acc0[nb], false, false);
      acc1[nb] = __builtin_amdgcn_wmma_f32_16x16x32_bf16(
          false, a1, false, b, (short)0, acc1[nb], false, false);
    }
  }
}

// C/D layout: lane l -> N=l&15, M = v + 8*(l>>4) within a 16-row subtile.
__device__ __forceinline__ void store_tile_bias(float* __restrict__ out, int r0, int R,
                                                const float* __restrict__ bias,
                                                const v8f acc[8]) {
  const int lane = threadIdx.x & 31;
  const int mh = (lane >> 4) * 8, nl = lane & 15;
#pragma unroll
  for (int nb = 0; nb < 8; ++nb) {
    int n = nb * 16 + nl;
    float bn = bias[n];
#pragma unroll
    for (int vi = 0; vi < 8; ++vi) {
      int r = r0 + mh + vi;
      if (r < R) out[(size_t)r * H + n] = acc[nb][vi] + bn;
    }
  }
}

__device__ __forceinline__ void store_tile_lds_f32(float* lds, int m0,
                                                   const float* __restrict__ bias,
                                                   const v8f acc[8]) {
  const int lane = threadIdx.x & 31;
  const int mh = (lane >> 4) * 8, nl = lane & 15;
#pragma unroll
  for (int nb = 0; nb < 8; ++nb) {
    int n = nb * 16 + nl;
    float bn = bias[n];
#pragma unroll
    for (int vi = 0; vi < 8; ++vi)
      lds[(size_t)(m0 + mh + vi) * H + n] = acc[nb][vi] + bn;
  }
}

__device__ __forceinline__ void store_tile_lds_bf16(unsigned short* lds, int m0,
                                                    const float* __restrict__ bias,
                                                    const v8f acc[8]) {
  const int lane = threadIdx.x & 31;
  const int mh = (lane >> 4) * 8, nl = lane & 15;
#pragma unroll
  for (int nb = 0; nb < 8; ++nb) {
    int n = nb * 16 + nl;
    float bn = bias[n];
#pragma unroll
    for (int vi = 0; vi < 8; ++vi)
      lds[(size_t)(m0 + mh + vi) * H + n] = f2bf(acc[nb][vi] + bn);
  }
}

// ---- small utility kernels -------------------------------------------------
__global__ void k_wt_bf16(const float* __restrict__ W, unsigned short* __restrict__ WT,
                          int K, int Kpad) {
  int i = blockIdx.x * 256 + threadIdx.x;
  if (i >= 128 * Kpad) return;
  int n = i / Kpad, k = i % Kpad;
  WT[i] = f2bf(k < K ? W[(size_t)k * H + n] : 0.0f);
}

__global__ void k_coord2diff(const float* __restrict__ x, const float* __restrict__ eattr,
                             const int* __restrict__ rowI, const int* __restrict__ colI,
                             float* __restrict__ ea, float* __restrict__ cdf, int E) {
  int e = blockIdx.x * 256 + threadIdx.x;
  if (e >= E) return;
  int r = rowI[e], c = colI[e];
  float dx = x[c * 3 + 0] - x[r * 3 + 0];
  float dy = x[c * 3 + 1] - x[r * 3 + 1];
  float dz = x[c * 3 + 2] - x[r * 3 + 2];
  float radial = dx * dx + dy * dy + dz * dz;
  float nrm = sqrtf(radial + 1e-8f) + 1.0f;   // NORM_CONSTANT
  cdf[e * 3 + 0] = dx / nrm;
  cdf[e * 3 + 1] = dy / nrm;
  cdf[e * 3 + 2] = dz / nrm;
  ea[e * 2 + 0] = radial;
  ea[e * 2 + 1] = eattr[e];
}

__global__ __launch_bounds__(128) void k_bn_stats(const float* __restrict__ src,
                                                  float* __restrict__ stats, int E) {
  const int t = threadIdx.x;               // column
  int e0 = blockIdx.x * 512;
  int e1 = e0 + 512; if (e1 > E) e1 = E;
  float s = 0.f, s2 = 0.f;
  for (int e = e0; e < e1; ++e) {
    __builtin_prefetch(&src[(size_t)(e + 32) * H + t], 0, 1);  // global_prefetch_b8
    float v = src[(size_t)e * H + t];
    s += v; s2 += v * v;
  }
  atomicAdd(&stats[t], s);
  atomicAdd(&stats[128 + t], s2);
}

__global__ void k_bn_finalize(const float* __restrict__ stats, const float* __restrict__ g,
                              const float* __restrict__ b, float* __restrict__ scale,
                              float* __restrict__ shift, float cnt) {
  int t = threadIdx.x;
  float invE = 1.0f / cnt;
  float mean = stats[t] * invE;
  float var  = stats[128 + t] * invE - mean * mean;   // biased var (jnp.var)
  float sc   = g[t] * rsqrtf(var + 1e-5f);
  scale[t] = sc;
  shift[t] = b[t] - mean * sc;
}

// ---- GEMM 1: inp = [h[col] | h[row] | ea] (K=258 pad 288) @ W1 + b1 -------
__global__ __launch_bounds__(128) void k_edge_gemm1(
    const float* __restrict__ hcur, const float* __restrict__ ea,
    const int* __restrict__ rowI, const int* __restrict__ colI,
    const unsigned short* __restrict__ WT, const float* __restrict__ bias,
    float* __restrict__ out, int E)
{
  extern __shared__ unsigned char smem[];
  unsigned short* ldsA  = (unsigned short*)smem;                   // 128*288
  unsigned short* ldsWT = (unsigned short*)(smem + 128 * 288 * 2); // 128*288
  const int t = threadIdx.x;
  {
    const uint32_t* s = (const uint32_t*)WT;
    uint32_t* d = (uint32_t*)ldsWT;
    for (int i = t; i < 128 * 288 / 2; i += 128) d[i] = s[i];
  }
  const int e0 = blockIdx.x * 128;
  for (int i = t; i < 128 * 288; i += 128) {
    int el = i / 288, k = i % 288;
    int e = e0 + el;
    float v = 0.0f;
    if (e < E) {
      if (k < 128)       v = hcur[(size_t)colI[e] * H + k];
      else if (k < 256)  v = hcur[(size_t)rowI[e] * H + (k - 128)];
      else if (k < 258)  v = ea[(size_t)e * 2 + (k - 256)];
    }
    ldsA[i] = f2bf(v);
  }
  __syncthreads();
  const int m0 = (t >> 5) * 32;
  v8f acc0[8], acc1[8];
  zero_acc(acc0); zero_acc(acc1);
  wmma_rowtile2<9>(ldsA, 288, ldsWT, m0, acc0, acc1);
  store_tile_bias(out, e0 + m0,      E, bias, acc0);
  store_tile_bias(out, e0 + m0 + 16, E, bias, acc1);
}

// ---- GEMM 2 + sigmoid attention + segment_sum scatter ---------------------
__global__ __launch_bounds__(128) void k_edge_gemm2_att(
    const float* __restrict__ src, const float* __restrict__ scale,
    const float* __restrict__ shift,
    const unsigned short* __restrict__ WT, const float* __restrict__ bias,
    const float* __restrict__ attw, const float* __restrict__ attb,
    const int* __restrict__ colI, float* __restrict__ agg, int E)
{
  extern __shared__ unsigned char smem[];
  unsigned short* ldsA   = (unsigned short*)smem;                               // 128*128
  unsigned short* ldsWT  = (unsigned short*)(smem + 128 * 128 * 2);             // 128*128
  float*          ldsOut = (float*)(smem + 128 * 128 * 2 + 128 * 128 * 2);      // 128*128 f32
  float*          ldsS   = (float*)(smem + 128 * 128 * 2 * 2 + 128 * 128 * 4);  // 128
  const int t = threadIdx.x;
  {
    const uint32_t* s = (const uint32_t*)WT;
    uint32_t* d = (uint32_t*)ldsWT;
    for (int i = t; i < 128 * 128 / 2; i += 128) d[i] = s[i];
  }
  const int e0 = blockIdx.x * 128;
  for (int i = t; i < 128 * 128; i += 128) {
    int el = i / 128, k = i % 128;
    int e = e0 + el;
    float v = 0.0f;
    if (e < E) v = src[(size_t)e * H + k] * scale[k] + shift[k];
    ldsA[i] = f2bf(v);
  }
  __syncthreads();
  const int m0 = (t >> 5) * 32;
  v8f acc0[8], acc1[8];
  zero_acc(acc0); zero_acc(acc1);
  wmma_rowtile2<4>(ldsA, 128, ldsWT, m0, acc0, acc1);
  store_tile_lds_f32(ldsOut, m0,      bias, acc0);
  store_tile_lds_f32(ldsOut, m0 + 16, bias, acc1);
  __syncthreads();
  {
    float s = 0.0f;
    int e = e0 + t;
    if (e < E) {
      float dot = attb[0];
      const float* r = ldsOut + (size_t)t * H;
#pragma unroll 4
      for (int j = 0; j < H; ++j) dot += r[j] * attw[j];
      s = sigmoidf_(dot) * 0.01f;   // fold 1/NORM_FACTOR
    }
    ldsS[t] = s;
  }
  __syncthreads();
  for (int el = 0; el < 128; ++el) {
    int e = e0 + el;
    if (e < E) {
      float v = ldsOut[(size_t)el * H + t] * ldsS[el];
      atomicAdd(&agg[(size_t)colI[e] * H + t], v);
    }
  }
}

// ---- GEMM with BN-affine input, plain f32 output (may alias src) ----------
__global__ __launch_bounds__(128) void k_edge_gemm_bnin(
    const float* __restrict__ src, const float* __restrict__ scale,
    const float* __restrict__ shift,
    const unsigned short* __restrict__ WT, const float* __restrict__ bias,
    float* __restrict__ dst, int E)
{
  extern __shared__ unsigned char smem[];
  unsigned short* ldsA  = (unsigned short*)smem;                   // 128*128
  unsigned short* ldsWT = (unsigned short*)(smem + 128 * 128 * 2); // 128*128
  const int t = threadIdx.x;
  {
    const uint32_t* s = (const uint32_t*)WT;
    uint32_t* d = (uint32_t*)ldsWT;
    for (int i = t; i < 128 * 128 / 2; i += 128) d[i] = s[i];
  }
  const int e0 = blockIdx.x * 128;
  for (int i = t; i < 128 * 128; i += 128) {
    int el = i / 128, k = i % 128;
    int e = e0 + el;
    float v = 0.0f;
    if (e < E) v = src[(size_t)e * H + k] * scale[k] + shift[k];
    ldsA[i] = f2bf(v);
  }
  __syncthreads();
  const int m0 = (t >> 5) * 32;
  v8f acc0[8], acc1[8];
  zero_acc(acc0); zero_acc(acc1);
  wmma_rowtile2<4>(ldsA, 128, ldsWT, m0, acc0, acc1);
  store_tile_bias(dst, e0 + m0,      E, bias, acc0);
  store_tile_bias(dst, e0 + m0 + 16, E, bias, acc1);
}

// ---- fused node MLP: h += W2*(W1*[h|agg]+b1)+b2 ---------------------------
__global__ __launch_bounds__(128) void k_node_mlp(
    const float* __restrict__ agg,
    const unsigned short* __restrict__ WT1, const float* __restrict__ b1,
    const unsigned short* __restrict__ WT2, const float* __restrict__ b2,
    float* __restrict__ hcur, int N)
{
  extern __shared__ unsigned char smem[];
  unsigned short* ldsA   = (unsigned short*)smem;                                        // 128*256
  unsigned short* ldsWT1 = (unsigned short*)(smem + 128 * 256 * 2);                      // 128*256
  unsigned short* ldsM   = (unsigned short*)(smem + 128 * 256 * 2 * 2);                  // 128*128
  unsigned short* ldsWT2 = (unsigned short*)(smem + 128 * 256 * 2 * 2 + 128 * 128 * 2);  // 128*128
  const int t = threadIdx.x;
  {
    const uint32_t* s1 = (const uint32_t*)WT1;
    uint32_t* d1 = (uint32_t*)ldsWT1;
    for (int i = t; i < 128 * 256 / 2; i += 128) d1[i] = s1[i];
    const uint32_t* s2 = (const uint32_t*)WT2;
    uint32_t* d2 = (uint32_t*)ldsWT2;
    for (int i = t; i < 128 * 128 / 2; i += 128) d2[i] = s2[i];
  }
  const int n0 = blockIdx.x * 128;
  for (int i = t; i < 128 * 256; i += 128) {
    int nl_ = i / 256, k = i % 256;
    int node = n0 + nl_;
    float v = 0.0f;
    if (node < N) {
      if (k < 128) v = hcur[(size_t)node * H + k];
      else         v = agg[(size_t)node * H + (k - 128)];
    }
    ldsA[i] = f2bf(v);
  }
  __syncthreads();
  const int m0 = (t >> 5) * 32;
  {
    v8f acc0[8], acc1[8];
    zero_acc(acc0); zero_acc(acc1);
    wmma_rowtile2<8>(ldsA, 256, ldsWT1, m0, acc0, acc1);
    store_tile_lds_bf16(ldsM, m0,      b1, acc0);
    store_tile_lds_bf16(ldsM, m0 + 16, b1, acc1);
  }
  __syncthreads();
  {
    v8f acc0[8], acc1[8];
    zero_acc(acc0); zero_acc(acc1);
    wmma_rowtile2<4>(ldsM, 128, ldsWT2, m0, acc0, acc1);
    const int lane = t & 31;
    const int mh = (lane >> 4) * 8, nl = lane & 15;
#pragma unroll
    for (int sub = 0; sub < 2; ++sub) {
      const v8f* acc = sub ? acc1 : acc0;
#pragma unroll
      for (int nb = 0; nb < 8; ++nb) {
        int n = nb * 16 + nl;
        float bn = b2[n];
#pragma unroll
        for (int vi = 0; vi < 8; ++vi) {
          int node = n0 + m0 + sub * 16 + mh + vi;
          if (node < N) {
            size_t idx = (size_t)node * H + n;
            hcur[idx] = hcur[idx] + acc[nb][vi] + bn;   // residual
          }
        }
      }
    }
  }
}

// ---- equivariant epilogue: t = BN2(y) @ w3; scatter coord updates ---------
__global__ void k_equiv_final(const float* __restrict__ src, const float* __restrict__ scale,
                              const float* __restrict__ shift, const float* __restrict__ w3,
                              const float* __restrict__ cdf, const int* __restrict__ colI,
                              float* __restrict__ aggx, int E) {
  int e = blockIdx.x * 256 + threadIdx.x;
  if (e >= E) return;
  float tsum = 0.0f;
  const float* r = src + (size_t)e * H;
#pragma unroll 4
  for (int j = 0; j < H; ++j) tsum += (r[j] * scale[j] + shift[j]) * w3[j];
  tsum *= 0.01f;   // 1/NORM_FACTOR
  int c = colI[e];
  atomicAdd(&aggx[c * 3 + 0], cdf[e * 3 + 0] * tsum);
  atomicAdd(&aggx[c * 3 + 1], cdf[e * 3 + 1] * tsum);
  atomicAdd(&aggx[c * 3 + 2], cdf[e * 3 + 2] * tsum);
}

__global__ void k_finalize(const float* __restrict__ hcur, const float* __restrict__ x,
                           const float* __restrict__ aggx, const float* __restrict__ ligand,
                           float* __restrict__ out, int N) {
  int i = blockIdx.x * 256 + threadIdx.x;
  int total = N * H + N * 3;
  if (i >= total) return;
  if (i < N * H) {
    out[i] = hcur[i];
  } else {
    int j = i - N * H;
    out[i] = x[j] + aggx[j] * ligand[j / 3];
  }
}

// ---------------------------------------------------------------------------
extern "C" void kernel_launch(void* const* d_in, const int* in_sizes, int n_in,
                              void* d_out, int out_size, void* d_ws, size_t ws_size,
                              hipStream_t stream)
{
  (void)in_sizes; (void)n_in; (void)out_size; (void)ws_size;
  const int E = NEDGES, N = NNODES;
  const float* h      = (const float*)d_in[0];
  const float* x      = (const float*)d_in[1];
  const float* eattr  = (const float*)d_in[2];
  const float* ligand = (const float*)d_in[3];
  const int*   eidx   = (const int*)d_in[4];
  const int* rowI = eidx;       // edge_index[0]
  const int* colI = eidx + E;   // edge_index[1]

  uint8_t* ws = (uint8_t*)d_ws;
  size_t off = 0;
  auto alloc = [&](size_t bytes) -> void* {
    void* p = ws + off;
    off += (bytes + 255) & ~(size_t)255;
    return p;
  };
  float* ebuf  = (float*)alloc((size_t)E * H * 4);   // 409.6 MB
  float* ea    = (float*)alloc((size_t)E * 2 * 4);
  float* cdf   = (float*)alloc((size_t)E * 3 * 4);
  float* agg   = (float*)alloc((size_t)N * H * 4);
  float* aggx  = (float*)alloc((size_t)N * 3 * 4);
  float* hcur  = (float*)alloc((size_t)N * H * 4);
  float* stats = (float*)alloc(256 * 4);
  float* bsc   = (float*)alloc(128 * 4);
  float* bsh   = (float*)alloc(128 * 4);
  unsigned short* wt_ew1[2]; unsigned short* wt_ew2[2];
  unsigned short* wt_n1[2];  unsigned short* wt_n2[2];
  for (int L = 0; L < 2; ++L) {
    wt_ew1[L] = (unsigned short*)alloc(288 * 128 * 2);
    wt_ew2[L] = (unsigned short*)alloc(128 * 128 * 2);
    wt_n1[L]  = (unsigned short*)alloc(256 * 128 * 2);
    wt_n2[L]  = (unsigned short*)alloc(128 * 128 * 2);
  }
  unsigned short* wt_qw1 = (unsigned short*)alloc(288 * 128 * 2);
  unsigned short* wt_qw2 = (unsigned short*)alloc(128 * 128 * 2);

  const size_t SM_G1   = 128 * 288 * 2 * 2;                                        // 147456
  const size_t SM_G2A  = 128 * 128 * 2 * 2 + 128 * 128 * 4 + 128 * 4;              // 131584
  const size_t SM_BNIN = 128 * 128 * 2 * 2;                                        //  65536
  const size_t SM_NODE = 128 * 256 * 2 * 2 + 128 * 128 * 2 * 2;                    // 196608

  // weights -> bf16, transposed [N][Kpad], K zero-padded
  for (int L = 0; L < 2; ++L) {
    int pb = 5 + 12 * L;
    k_wt_bf16<<<(128 * 288 + 255) / 256, 256, 0, stream>>>((const float*)d_in[pb + 0], wt_ew1[L], 258, 288);
    k_wt_bf16<<<(128 * 128 + 255) / 256, 256, 0, stream>>>((const float*)d_in[pb + 4], wt_ew2[L], 128, 128);
    k_wt_bf16<<<(128 * 256 + 255) / 256, 256, 0, stream>>>((const float*)d_in[pb + 8], wt_n1[L], 256, 256);
    k_wt_bf16<<<(128 * 128 + 255) / 256, 256, 0, stream>>>((const float*)d_in[pb + 10], wt_n2[L], 128, 128);
  }
  k_wt_bf16<<<(128 * 288 + 255) / 256, 256, 0, stream>>>((const float*)d_in[29], wt_qw1, 258, 288);
  k_wt_bf16<<<(128 * 128 + 255) / 256, 256, 0, stream>>>((const float*)d_in[33], wt_qw2, 128, 128);

  hipMemcpyAsync(hcur, h, (size_t)N * H * 4, hipMemcpyDeviceToDevice, stream);
  k_coord2diff<<<(E + 255) / 256, 256, 0, stream>>>(x, eattr, rowI, colI, ea, cdf, E);

  const int EG = (E + 127) / 128, NG = (N + 127) / 128;

  for (int L = 0; L < 2; ++L) {
    int pb = 5 + 12 * L;
    const float* e_b1 = (const float*)d_in[pb + 1];
    const float* e_g  = (const float*)d_in[pb + 2];
    const float* e_bb = (const float*)d_in[pb + 3];
    const float* e_b2 = (const float*)d_in[pb + 5];
    const float* attw = (const float*)d_in[pb + 6];
    const float* attb = (const float*)d_in[pb + 7];
    const float* n_b1 = (const float*)d_in[pb + 9];
    const float* n_b2 = (const float*)d_in[pb + 11];

    k_edge_gemm1<<<EG, 128, SM_G1, stream>>>(hcur, ea, rowI, colI, wt_ew1[L], e_b1, ebuf, E);
    hipMemsetAsync(stats, 0, 256 * 4, stream);
    k_bn_stats<<<(E + 511) / 512, 128, 0, stream>>>(ebuf, stats, E);
    k_bn_finalize<<<1, 128, 0, stream>>>(stats, e_g, e_bb, bsc, bsh, (float)E);
    hipMemsetAsync(agg, 0, (size_t)N * H * 4, stream);
    k_edge_gemm2_att<<<EG, 128, SM_G2A, stream>>>(ebuf, bsc, bsh, wt_ew2[L], e_b2,
                                                  attw, attb, colI, agg, E);
    k_node_mlp<<<NG, 128, SM_NODE, stream>>>(agg, wt_n1[L], n_b1, wt_n2[L], n_b2, hcur, N);
  }

  // EquivariantUpdate
  const float* q_b1 = (const float*)d_in[30];
  const float* q_g1 = (const float*)d_in[31];
  const float* q_bb1 = (const float*)d_in[32];
  const float* q_b2 = (const float*)d_in[34];
  const float* q_g2 = (const float*)d_in[35];
  const float* q_bb2 = (const float*)d_in[36];
  const float* q_w3 = (const float*)d_in[37];

  k_edge_gemm1<<<EG, 128, SM_G1, stream>>>(hcur, ea, rowI, colI, wt_qw1, q_b1, ebuf, E);
  hipMemsetAsync(stats, 0, 256 * 4, stream);
  k_bn_stats<<<(E + 511) / 512, 128, 0, stream>>>(ebuf, stats, E);
  k_bn_finalize<<<1, 128, 0, stream>>>(stats, q_g1, q_bb1, bsc, bsh, (float)E);
  k_edge_gemm_bnin<<<EG, 128, SM_BNIN, stream>>>(ebuf, bsc, bsh, wt_qw2, q_b2, ebuf, E);
  hipMemsetAsync(stats, 0, 256 * 4, stream);
  k_bn_stats<<<(E + 511) / 512, 128, 0, stream>>>(ebuf, stats, E);
  k_bn_finalize<<<1, 128, 0, stream>>>(stats, q_g2, q_bb2, bsc, bsh, (float)E);
  hipMemsetAsync(aggx, 0, (size_t)N * 3 * 4, stream);
  k_equiv_final<<<(E + 255) / 256, 256, 0, stream>>>(ebuf, bsc, bsh, q_w3, cdf, colI, aggx, E);
  k_finalize<<<(N * (H + 3) + 255) / 256, 256, 0, stream>>>(hcur, x, aggx, ligand,
                                                            (float*)d_out, N);
}
